// ODERNN_85718957293961
// MI455X (gfx1250) — matrix operand order repaired and checked by reading, not compiled, not run on hardware
//
#include <hip/hip_runtime.h>
#include <math.h>

#define B_  1024
#define T_  100
#define D_  64
#define H_  512
#define HID 50
#define HIDP 64
#define H2  256
#define NPART 256
#define ATOL_ 0.1f
#define RTOL_ 0.1f
#define ROWS 32            // rows per block in fused ODE-f kernel
#define ALD  (H_ + 8)      // padded LDS stride (bf16 elems) for A
#define ZLD  (HIDP + 8)    // padded LDS stride for Z

// scalar slots in workspace
#define S_T    0
#define S_H    1
#define S_HTRY 2
#define S_DONE 3
#define S_SEL  4
#define S_H0   5
#define S_D1   6

typedef __attribute__((ext_vector_type(16))) __bf16 v16bf;
typedef __attribute__((ext_vector_type(8)))  float  v8f;
typedef __attribute__((ext_vector_type(4)))  float  f32x4;
typedef __attribute__((ext_vector_type(4)))  unsigned int u32x4;

union BFrag {
    v16bf v;
    u32x4 q[2];
    unsigned int u[8];
};

__device__ __forceinline__ unsigned int pack2bf(float lo, float hi) {
    unsigned a = __float_as_uint(lo);
    a += 0x7fffu + ((a >> 16) & 1u);
    unsigned b = __float_as_uint(hi);
    b += 0x7fffu + ((b >> 16) & 1u);
    return (a >> 16) | (b & 0xffff0000u);
}

__device__ __forceinline__ unsigned short f2bf(float v) {
    unsigned a = __float_as_uint(v);
    a += 0x7fffu + ((a >> 16) & 1u);
    return (unsigned short)(a >> 16);
}

// 16x16 tile accumulate over K (mult of 32), A fp32 (packed in-register), W bf16.
__device__ __forceinline__ v8f wmma_tile_f32a(const float* __restrict__ A, long lda,
                                              const unsigned short* __restrict__ W, long ldw,
                                              int K, v8f acc, int lane) {
    const int mrow = lane & 15;
    const int hi   = lane >> 4;
    const float* arow = A + (long)mrow * lda + (hi ? 8 : 0);
    const unsigned short* wrow = W + (long)(lane & 15) * ldw + (hi ? 8 : 0);
    for (int k0 = 0; k0 < K; k0 += 32) {
        BFrag af, bf;
        const f32x4* ap0 = (const f32x4*)(arow + k0);
        const f32x4* ap1 = (const f32x4*)(arow + k0 + 16);
        f32x4 a0 = ap0[0], a1 = ap0[1];
        f32x4 a2 = ap1[0], a3 = ap1[1];
        af.u[0] = pack2bf(a0.x, a0.y); af.u[1] = pack2bf(a0.z, a0.w);
        af.u[2] = pack2bf(a1.x, a1.y); af.u[3] = pack2bf(a1.z, a1.w);
        af.u[4] = pack2bf(a2.x, a2.y); af.u[5] = pack2bf(a2.z, a2.w);
        af.u[6] = pack2bf(a3.x, a3.y); af.u[7] = pack2bf(a3.z, a3.w);
        const u32x4* wp = (const u32x4*)(wrow + k0);
        bf.q[0] = wp[0]; bf.q[1] = wp[2];
        acc = __builtin_amdgcn_wmma_f32_16x16x32_bf16(false, af.v, false, bf.v,
                                                      (short)0, acc, false, false);
    }
    return acc;
}

// ===========================================================================
// Fused ODE function eval: kout = tanh(tanh(A W1^T + b1) W2^T + b2) * rowscale
// with A = y + hs*sum(ci*pi) built in LDS; optionally writes A fp32 (combo_out).
// One block = 32 batch rows.
// ===========================================================================
__global__ void __launch_bounds__(256)
fused_f(const float* __restrict__ y,
        const float* p1, float c1, const float* p2, float c2,
        const float* p3, float c3, const float* p4, float c4,
        const float* p5, float c5,
        const float* __restrict__ scal, int hsIdx,
        const unsigned short* __restrict__ W1p, const float* __restrict__ b1,
        const unsigned short* __restrict__ W2p, const float* __restrict__ b2,
        const float* __restrict__ dt, int tstep, int Tsteps,
        float* __restrict__ kout, float* __restrict__ combo_out) {
    __shared__ unsigned short Abuf[ROWS * ALD];
    __shared__ unsigned short Zbuf[ROWS * ZLD];

    const int r0 = blockIdx.x * ROWS;
    const float hs = (hsIdx >= 0) ? scal[hsIdx] : 1.0f;

    // ---- cooperative combo + convert into LDS (optionally persist fp32) ----
    for (int i = threadIdx.x; i < ROWS * H_; i += 256) {
        int r = i >> 9;            // / H_
        int c = i & (H_ - 1);
        long gi = (long)(r0 + r) * H_ + c;
        float v = y[gi];
        float s = 0.f;
        if (p1) s += c1 * p1[gi];
        if (p2) s += c2 * p2[gi];
        if (p3) s += c3 * p3[gi];
        if (p4) s += c4 * p4[gi];
        if (p5) s += c5 * p5[gi];
        v += hs * s;
        if (combo_out) combo_out[gi] = v;
        Abuf[r * ALD + c] = f2bf(v);
    }
    __syncthreads();

    const int wave = threadIdx.x >> 5;
    const int lane = threadIdx.x & 31;
    const int mt = wave & 1;           // which 16-row half
    const int wq = wave >> 1;          // 0..3
    const int hi = lane >> 4;
    const int mrow = mt * 16 + (lane & 15);

    // ---- stage 1: Z(32x64) = tanh(A @ W1p^T + b1), into LDS bf16 ----
    {
        v8f acc = {};
        const unsigned short* arow = Abuf + mrow * ALD + (hi ? 8 : 0);
        const unsigned short* wrow = W1p + (long)(wq * 16 + (lane & 15)) * H_ + (hi ? 8 : 0);
        for (int k0 = 0; k0 < H_; k0 += 32) {
            BFrag af, bf;
            af.q[0] = *(const u32x4*)(arow + k0);
            af.q[1] = *(const u32x4*)(arow + k0 + 16);
            bf.q[0] = *(const u32x4*)(wrow + k0);
            bf.q[1] = *(const u32x4*)(wrow + k0 + 16);
            acc = __builtin_amdgcn_wmma_f32_16x16x32_bf16(false, af.v, false, bf.v,
                                                          (short)0, acc, false, false);
        }
        int ccol = wq * 16 + (lane & 15);
        float bv = (ccol < HID) ? b1[ccol] : 0.f;
        int rb = mt * 16 + (hi ? 8 : 0);
        #pragma unroll
        for (int r = 0; r < 8; ++r)
            Zbuf[(rb + r) * ZLD + ccol] = f2bf(tanhf(acc[r] + bv));
    }
    __syncthreads();

    // ---- stage 2: kout(32x512) = tanh(Z @ W2p^T + b2) * rowscale ----
    {
        const unsigned short* arow = Zbuf + mrow * ZLD + (hi ? 8 : 0);
        BFrag af0, af1;
        af0.q[0] = *(const u32x4*)(arow + 0);
        af0.q[1] = *(const u32x4*)(arow + 16);
        af1.q[0] = *(const u32x4*)(arow + 32);
        af1.q[1] = *(const u32x4*)(arow + 48);

        int rb = mt * 16 + (hi ? 8 : 0);
        float rs[8];
        #pragma unroll
        for (int r = 0; r < 8; ++r) {
            const float* dp = dt + (long)(r0 + rb + r) * Tsteps * 2 + (long)tstep * 2;
            rs[r] = (dp[1] - dp[0]) * 0.01f;
        }

        #pragma unroll
        for (int j = 0; j < 8; ++j) {
            int nt = wq + j * 4;   // 0..31 column tiles
            v8f acc = {};
            const unsigned short* wrow = W2p + (long)(nt * 16 + (lane & 15)) * HIDP + (hi ? 8 : 0);
            BFrag bf;
            bf.q[0] = *(const u32x4*)(wrow + 0);
            bf.q[1] = *(const u32x4*)(wrow + 16);
            acc = __builtin_amdgcn_wmma_f32_16x16x32_bf16(false, af0.v, false, bf.v,
                                                          (short)0, acc, false, false);
            bf.q[0] = *(const u32x4*)(wrow + 32);
            bf.q[1] = *(const u32x4*)(wrow + 48);
            acc = __builtin_amdgcn_wmma_f32_16x16x32_bf16(false, af1.v, false, bf.v,
                                                          (short)0, acc, false, false);
            int ccol = nt * 16 + (lane & 15);
            float bv = b2[ccol];
            #pragma unroll
            for (int r = 0; r < 8; ++r) {
                long grow = r0 + rb + r;
                kout[grow * H_ + ccol] = tanhf(acc[r] + bv) * rs[r];
            }
        }
    }
}

// ===========================================================================
// RNN cell: C = tanh(X*Wih^T + Hp*Whh^T + bih + bhh)    (fp32 A, in-reg pack)
// ===========================================================================
__global__ void __launch_bounds__(256)
wmma_rnn(const float* __restrict__ X, long ldx,
         const unsigned short* __restrict__ Wih,
         const float* __restrict__ Hp,
         const unsigned short* __restrict__ Whh,
         float* __restrict__ C,
         const float* __restrict__ bih, const float* __restrict__ bhh,
         int M, int N, int Kx, int Kh) {
    int wave = blockIdx.x * (blockDim.x >> 5) + (threadIdx.x >> 5);
    int ntN = N >> 4;
    int ntiles = (M >> 4) * ntN;
    if (wave >= ntiles) return;
    int tileM = (wave / ntN) << 4;
    int tileN = (wave % ntN) << 4;
    int lane = threadIdx.x & 31;

    v8f acc = {};
    acc = wmma_tile_f32a(X + (long)tileM * ldx, ldx, Wih + (long)tileN * Kx, Kx, Kx, acc, lane);
    acc = wmma_tile_f32a(Hp + (long)tileM * Kh, Kh, Whh + (long)tileN * Kh, Kh, Kh, acc, lane);

    int ccol = tileN + (lane & 15);
    float bv = bih[ccol] + bhh[ccol];
    int rbase = tileM + ((lane >> 4) ? 8 : 0);
    for (int r = 0; r < 8; ++r)
        C[(long)(rbase + r) * N + ccol] = tanhf(acc[r] + bv);
}

// ===========================================================================
// Fused head: mu = relu(y Wl1^T + bl1) (WMMA -> LDS), out = mu.Wmu + bmu
// One block = 16 batch rows (one M tile), 8 waves cover the 16 N tiles.
// ===========================================================================
__global__ void __launch_bounds__(256)
head_fused(const float* __restrict__ y,
           const unsigned short* __restrict__ Wl1b, const float* __restrict__ bl1,
           const float* __restrict__ Wmu, const float* __restrict__ bmu,
           float* __restrict__ out, int tstep, int Tsteps) {
    __shared__ float muL[16 * H2];
    const int r0 = blockIdx.x * 16;
    const int wave = threadIdx.x >> 5;
    const int lane = threadIdx.x & 31;

    #pragma unroll
    for (int jj = 0; jj < 2; ++jj) {
        int nt = wave * 2 + jj;
        v8f acc = {};
        acc = wmma_tile_f32a(y + (long)r0 * H_, H_, Wl1b + (long)nt * 16 * H_, H_, H_, acc, lane);
        int ccol = nt * 16 + (lane & 15);
        float bv = bl1[ccol];
        int rb = (lane >> 4) ? 8 : 0;
        #pragma unroll
        for (int r = 0; r < 8; ++r)
            muL[(rb + r) * H2 + ccol] = fmaxf(acc[r] + bv, 0.f);
    }
    __syncthreads();

    #pragma unroll
    for (int jj = 0; jj < 2; ++jj) {
        int rr = wave * 2 + jj;
        float s = 0.f;
        for (int c = lane; c < H2; c += 32) s += muL[rr * H2 + c] * Wmu[c];
        for (int o = 16; o > 0; o >>= 1) s += __shfl_xor(s, o, 32);
        if (lane == 0) out[(long)(r0 + rr) * Tsteps + tstep] = s + bmu[0];
    }
}

// fp32 -> bf16 with zero padding
__global__ void cvt_bf16_pad(const float* __restrict__ src, unsigned short* __restrict__ dst,
                             int dR, int dC, int sR, int sC) {
    int i = blockIdx.x * blockDim.x + threadIdx.x;
    if (i >= dR * dC) return;
    int r = i / dC, c = i % dC;
    float v = (r < sR && c < sC) ? src[r * sC + c] : 0.f;
    dst[i] = f2bf(v);
}

__global__ void zero_kernel(float* p, int n) {
    int i = blockIdx.x * blockDim.x + threadIdx.x;
    if (i < n) p[i] = 0.f;
}

__global__ void commit_kernel(float* __restrict__ y, const float* __restrict__ y5,
                              const float* __restrict__ scal, int n) {
    int i = blockIdx.x * blockDim.x + threadIdx.x;
    if (i >= n) return;
    if (scal[S_SEL] != 0.f) y[i] = y5[i];
}

// mode 0: p0 += (a/sc)^2, p1 += (b/sc)^2, sc = ATOL+RTOL*|a|
// mode 1: p0 += ((b-c)/sc)^2, sc = ATOL+RTOL*|a|
__global__ void red_partial(int mode, const float* __restrict__ a, const float* __restrict__ b,
                            const float* __restrict__ c,
                            float* __restrict__ part0, float* __restrict__ part1, int n) {
    __shared__ float s0[256], s1[256];
    float acc0 = 0.f, acc1 = 0.f;
    for (int i = blockIdx.x * blockDim.x + threadIdx.x; i < n; i += gridDim.x * blockDim.x) {
        if (mode == 0) {
            float sc = ATOL_ + RTOL_ * fabsf(a[i]);
            float u = a[i] / sc, v = b[i] / sc;
            acc0 += u * u; acc1 += v * v;
        } else {
            float sc = ATOL_ + RTOL_ * fabsf(a[i]);
            float u = (b[i] - c[i]) / sc;
            acc0 += u * u;
        }
    }
    s0[threadIdx.x] = acc0; s1[threadIdx.x] = acc1;
    __syncthreads();
    for (int o = 128; o > 0; o >>= 1) {
        if (threadIdx.x < o) { s0[threadIdx.x] += s0[threadIdx.x + o]; s1[threadIdx.x] += s1[threadIdx.x + o]; }
        __syncthreads();
    }
    if (threadIdx.x == 0) { part0[blockIdx.x] = s0[0]; if (part1) part1[blockIdx.x] = s1[0]; }
}

// err formed on the fly: err = hs*(e1 k1 + e3 k3 + e4 k4 + e5 k5 + e6 k6 + e7 k7)
// sc = ATOL + RTOL*max(|y|,|y5|);  partial sums of (err/sc)^2
__global__ void red_err(const float* __restrict__ y, const float* __restrict__ y5,
                        const float* __restrict__ k1, float e1,
                        const float* __restrict__ k3, float e3,
                        const float* __restrict__ k4, float e4,
                        const float* __restrict__ k5, float e5,
                        const float* __restrict__ k6, float e6,
                        const float* __restrict__ k7, float e7,
                        const float* __restrict__ scal,
                        float* __restrict__ part0, int n) {
    __shared__ float s0[256];
    const float hs = scal[S_HTRY];
    float acc0 = 0.f;
    for (int i = blockIdx.x * blockDim.x + threadIdx.x; i < n; i += gridDim.x * blockDim.x) {
        float e = e1 * k1[i] + e3 * k3[i] + e4 * k4[i] + e5 * k5[i] + e6 * k6[i] + e7 * k7[i];
        e *= hs;
        float sc = ATOL_ + RTOL_ * fmaxf(fabsf(y[i]), fabsf(y5[i]));
        float u = e / sc;
        acc0 += u * u;
    }
    s0[threadIdx.x] = acc0;
    __syncthreads();
    for (int o = 128; o > 0; o >>= 1) {
        if (threadIdx.x < o) s0[threadIdx.x] += s0[threadIdx.x + o];
        __syncthreads();
    }
    if (threadIdx.x == 0) part0[blockIdx.x] = s0[0];
}

__global__ void fin_init1(float* __restrict__ scal, const float* __restrict__ part0,
                          const float* __restrict__ part1, int n) {
    __shared__ float s0[256], s1[256];
    s0[threadIdx.x] = part0[threadIdx.x];
    s1[threadIdx.x] = part1[threadIdx.x];
    __syncthreads();
    for (int o = 128; o > 0; o >>= 1) {
        if (threadIdx.x < o) { s0[threadIdx.x] += s0[threadIdx.x + o]; s1[threadIdx.x] += s1[threadIdx.x + o]; }
        __syncthreads();
    }
    if (threadIdx.x == 0) {
        float d0 = sqrtf(s0[0] / (float)n);
        float d1 = sqrtf(s1[0] / (float)n);
        float h0 = (d0 < 1e-5f || d1 < 1e-5f) ? 1e-6f : 0.01f * d0 / fmaxf(d1, 1e-12f);
        scal[S_H0] = h0;
        scal[S_D1] = d1;
    }
}

__global__ void fin_init2(float* __restrict__ scal, const float* __restrict__ part0, int n) {
    __shared__ float s0[256];
    s0[threadIdx.x] = part0[threadIdx.x];
    __syncthreads();
    for (int o = 128; o > 0; o >>= 1) {
        if (threadIdx.x < o) s0[threadIdx.x] += s0[threadIdx.x + o];
        __syncthreads();
    }
    if (threadIdx.x == 0) {
        float h0 = scal[S_H0];
        float d2 = sqrtf(s0[0] / (float)n) / h0;
        float dmax = fmaxf(scal[S_D1], d2);
        float h1 = (dmax <= 1e-15f) ? fmaxf(1e-6f, h0 * 1e-3f)
                                    : powf(0.01f / fmaxf(dmax, 1e-15f), 0.2f);
        scal[S_H] = fminf(fminf(100.f * h0, h1), 1.0f);
        scal[S_T] = 0.f;
    }
}

__global__ void ctrl_pre(float* scal) {
    float t = scal[S_T], h = scal[S_H];
    scal[S_DONE] = (t >= 1.0f) ? 1.f : 0.f;
    scal[S_HTRY] = fminf(h, 1.0f - t);
}

__global__ void ctrl_post(float* __restrict__ scal, const float* __restrict__ part0, int n) {
    __shared__ float s0[256];
    s0[threadIdx.x] = part0[threadIdx.x];
    __syncthreads();
    for (int o = 128; o > 0; o >>= 1) {
        if (threadIdx.x < o) s0[threadIdx.x] += s0[threadIdx.x + o];
        __syncthreads();
    }
    if (threadIdx.x == 0) {
        float norm = sqrtf(s0[0] / (float)n);
        bool done = scal[S_DONE] != 0.f;
        bool accept = norm <= 1.0f;
        float factor = fminf(fmaxf(0.9f * powf(fmaxf(norm, 1e-10f), -0.2f), 0.2f), 10.0f);
        if (!done) {
            if (accept) scal[S_T] += scal[S_HTRY];
            scal[S_H] = scal[S_HTRY] * factor;
        }
        scal[S_SEL] = (!done && accept) ? 1.f : 0.f;
    }
}

extern "C" void kernel_launch(void* const* d_in, const int* in_sizes, int n_in,
                              void* d_out, int out_size, void* d_ws, size_t ws_size,
                              hipStream_t stream) {
    (void)in_sizes; (void)n_in; (void)out_size; (void)ws_size;
    const float* dt   = (const float*)d_in[0];
    const float* x    = (const float*)d_in[1];
    const float* W_ih = (const float*)d_in[2];
    const float* b_ih = (const float*)d_in[3];
    const float* W_hh = (const float*)d_in[4];
    const float* b_hh = (const float*)d_in[5];
    const float* W1   = (const float*)d_in[6];
    const float* b1   = (const float*)d_in[7];
    const float* W2   = (const float*)d_in[8];
    const float* b2   = (const float*)d_in[9];
    const float* Wl1  = (const float*)d_in[10];
    const float* bl1  = (const float*)d_in[11];
    const float* Wmu  = (const float*)d_in[12];
    const float* bmu  = (const float*)d_in[13];
    float* outp = (float*)d_out;

    char* ws = (char*)d_ws;
    size_t off = 0;
    auto alloc = [&](size_t bytes) -> void* {
        off = (off + 255) & ~(size_t)255;
        void* p = ws + off;
        off += bytes;
        return p;
    };
    const int n = B_ * H_;
    float* scal  = (float*)alloc(64 * sizeof(float));
    float* part0 = (float*)alloc(NPART * sizeof(float));
    float* part1 = (float*)alloc(NPART * sizeof(float));
    unsigned short* Wihb = (unsigned short*)alloc((size_t)H_ * D_ * 2);
    unsigned short* Whhb = (unsigned short*)alloc((size_t)H_ * H_ * 2);
    unsigned short* W1p  = (unsigned short*)alloc((size_t)HIDP * H_ * 2);
    unsigned short* W2p  = (unsigned short*)alloc((size_t)H_ * HIDP * 2);
    unsigned short* Wl1b = (unsigned short*)alloc((size_t)H2 * H_ * 2);
    float* hb0  = (float*)alloc((size_t)n * 4);
    float* hb1  = (float*)alloc((size_t)n * 4);
    float* y5b  = (float*)alloc((size_t)n * 4);
    float* kb[7];
    for (int i = 0; i < 7; ++i) kb[i] = (float*)alloc((size_t)n * 4);
    float* f0b = (float*)alloc((size_t)n * 4);
    float* f1b = (float*)alloc((size_t)n * 4);

    auto cvt = [&](const float* src, unsigned short* dst, int dR, int dC, int sR, int sC) {
        int tot = dR * dC;
        cvt_bf16_pad<<<(tot + 255) / 256, 256, 0, stream>>>(src, dst, dR, dC, sR, sC);
    };
    cvt(W_ih, Wihb, H_, D_, H_, D_);
    cvt(W_hh, Whhb, H_, H_, H_, H_);
    cvt(W1,   W1p,  HIDP, H_, HID, H_);
    cvt(W2,   W2p,  H_, HIDP, H_, HID);
    cvt(Wl1,  Wl1b, H2, H_, H2, H_);
    zero_kernel<<<n / 256, 256, 0, stream>>>(hb0, n);

    // fused f-eval: kout = f(y + hs*sum(ci*pi)); combo optionally persisted
    auto ff = [&](const float* yy, float* kout, int tstep, int hsIdx,
                  const float* q1 = nullptr, float d1 = 0.f,
                  const float* q2 = nullptr, float d2 = 0.f,
                  const float* q3 = nullptr, float d3 = 0.f,
                  const float* q4 = nullptr, float d4 = 0.f,
                  const float* q5 = nullptr, float d5 = 0.f,
                  float* combo_out = nullptr) {
        fused_f<<<B_ / ROWS, 256, 0, stream>>>(yy, q1, d1, q2, d2, q3, d3, q4, d4, q5, d5,
                                               scal, hsIdx, W1p, b1, W2p, b2, dt, tstep, T_,
                                               kout, combo_out);
    };

    const float e1 = 35.f/384.f - 5179.f/57600.f;
    const float e3 = 500.f/1113.f - 7571.f/16695.f;
    const float e4 = 125.f/192.f - 393.f/640.f;
    const float e5 = -2187.f/6784.f + 92097.f/339200.f;
    const float e6 = 11.f/84.f - 187.f/2100.f;
    const float e7 = -1.f/40.f;

    float* bufs[2] = { hb0, hb1 };
    for (int t = 0; t < T_; ++t) {
        const float* yprev = bufs[t & 1];
        float* y = bufs[(t + 1) & 1];

        // RNN cell
        {
            int tiles = (B_ / 16) * (H_ / 16);
            wmma_rnn<<<(tiles + 7) / 8, 256, 0, stream>>>(x + (long)t * D_, (long)T_ * D_, Wihb,
                                                          yprev, Whhb, y, b_ih, b_hh,
                                                          B_, H_, D_, H_);
        }

        // initial step size (batch-global scalar)
        ff(y, f0b, t, -1);
        red_partial<<<NPART, 256, 0, stream>>>(0, y, f0b, nullptr, part0, part1, n);
        fin_init1<<<1, 256, 0, stream>>>(scal, part0, part1, n);
        ff(y, f1b, t, S_H0, f0b, 1.f);
        red_partial<<<NPART, 256, 0, stream>>>(1, y, f1b, f0b, part0, part1, n);
        fin_init2<<<1, 256, 0, stream>>>(scal, part0, n);

        // 8 fixed controller iterations
        for (int it = 0; it < 8; ++it) {
            ctrl_pre<<<1, 1, 0, stream>>>(scal);
            ff(y, kb[0], t, -1);                                               // k1 = f(y)
            ff(y, kb[1], t, S_HTRY, kb[0], 0.2f);                              // k2
            ff(y, kb[2], t, S_HTRY, kb[0], 3.f/40.f, kb[1], 9.f/40.f);         // k3
            ff(y, kb[3], t, S_HTRY, kb[0], 44.f/45.f, kb[1], -56.f/15.f,
               kb[2], 32.f/9.f);                                               // k4
            ff(y, kb[4], t, S_HTRY, kb[0], 19372.f/6561.f, kb[1], -25360.f/2187.f,
               kb[2], 64448.f/6561.f, kb[3], -212.f/729.f);                    // k5
            ff(y, kb[5], t, S_HTRY, kb[0], 9017.f/3168.f, kb[1], -355.f/33.f,
               kb[2], 46732.f/5247.f, kb[3], 49.f/176.f, kb[4], -5103.f/18656.f); // k6
            // k7 = f(y5); y5 = y + h*(...) materialized as a byproduct
            ff(y, kb[6], t, S_HTRY, kb[0], 35.f/384.f, kb[2], 500.f/1113.f,
               kb[3], 125.f/192.f, kb[4], -2187.f/6784.f, kb[5], 11.f/84.f, y5b);
            red_err<<<NPART, 256, 0, stream>>>(y, y5b, kb[0], e1, kb[2], e3, kb[3], e4,
                                               kb[4], e5, kb[5], e6, kb[6], e7,
                                               scal, part0, n);
            ctrl_post<<<1, 256, 0, stream>>>(scal, part0, n);
            commit_kernel<<<n / 256, 256, 0, stream>>>(y, y5b, scal, n);
        }

        // fused output head
        head_fused<<<B_ / 16, 256, 0, stream>>>(y, Wl1b, bl1, Wmu, bmu, outp, t, T_);
    }
}
